// ExpandFormerV16_10496900071807
// MI455X (gfx1250) — compile-verified, matrix operand order checked by reading, not run on hardware
//
#include <hip/hip_runtime.h>
#include <cstdint>
#include <cstddef>

typedef __attribute__((ext_vector_type(16))) _Float16 v16h;
typedef __attribute__((ext_vector_type(8)))  _Float16 v8h;
typedef __attribute__((ext_vector_type(4)))  _Float16 v4h;
typedef __attribute__((ext_vector_type(8)))  float    v8f;

#define VOCAB 32000
#define DIM   512
#define ND    16
#define DD    128
#define HID   2048          // ND*DD
#define NTOK  16384         // B*S
#define MTILE 32            // tokens per workgroup (2 WMMA M-tiles)

// Padded LDS row strides (in halfs): +8 halfs = +16B -> 4-bank shift per row,
// so the 16 rows of an A-fragment load spread across all 64 LDS banks.
#define HSTR  (DIM + 8)     // 520 halfs  (1040 B, 16B-aligned)
#define ASTR  (HID + 8)     // 2056 halfs (4112 B, 16B-aligned)

// K-offset of element j (0..15) of a 16-bit WMMA A/B fragment for `lane`
// (ISA 7.12.2: lanes 0-15 hold K {0..7,16..23}, lanes 16-31 hold K {8..15,24..31})
__device__ __forceinline__ int frag_koff(int lane, int j) {
    return (j < 8 ? j : j + 8) + ((lane >= 16) ? 8 : 0);
}

// Branch-free tanh-form GELU: 0.5x(1+tanh(0.79788456(x+0.044715x^3)))
// tanh(u) = 1 - 2/(exp(2u)+1), exp via v_exp_f32 (2^x), rcp via v_rcp_f32.
__device__ __forceinline__ float gelu_fast(float v) {
    float u  = 0.7978845608028654f * (v + 0.044715f * v * v * v);
    float e  = __builtin_amdgcn_exp2f(u * 2.8853900817779268f);  // exp(2u)
    float th = 1.0f - 2.0f * __builtin_amdgcn_rcpf(e + 1.0f);
    return 0.5f * v * (1.0f + th);
}

// ---- Weight repack: W1 [16,512,128] f32 -> B-fragment-layout f16 ----
// Logical B1[K=512][N=2048], Ncol = n*128 + d.  Fragment (ntile 0..127, kt 0..15),
// packed half index = ((ntile*16 + kt)*512) + lane*16 + j.
__global__ void __launch_bounds__(256) pack_w1(const float* __restrict__ W1,
                                               _Float16* __restrict__ outp) {
    int idx = blockIdx.x * 256 + threadIdx.x;      // 0 .. 512*2048-1
    if (idx >= 512 * HID) return;
    int j     = idx & 15;
    int lane  = (idx >> 4) & 31;
    int frag  = idx >> 9;
    int kt    = frag & 15;
    int ntile = frag >> 4;
    int K     = kt * 32 + frag_koff(lane, j);
    int Ncol  = ntile * 16 + (lane & 15);
    int n     = Ncol >> 7;
    int dcol  = Ncol & 127;
    outp[idx] = (_Float16)W1[((size_t)n * DIM + K) * DD + dcol];
}

// ---- Weight repack: W2 [16,128,512] f32 -> B-fragment-layout f16 ----
// Logical B2[K=2048][N=512], Krow = n*128 + d.  Fragment (ntile2 0..31, kt 0..63),
// packed half index = ((ntile2*64 + kt)*512) + lane*16 + j.
__global__ void __launch_bounds__(256) pack_w2(const float* __restrict__ W2,
                                               _Float16* __restrict__ outp) {
    int idx = blockIdx.x * 256 + threadIdx.x;      // 0 .. 2048*512-1
    if (idx >= HID * DIM) return;
    int j      = idx & 15;
    int lane   = (idx >> 4) & 31;
    int frag   = idx >> 9;
    int kt     = frag & 63;
    int ntile2 = frag >> 6;
    int K      = kt * 32 + frag_koff(lane, j);
    int Dcol   = ntile2 * 16 + (lane & 15);
    int n      = K >> 7;
    int d      = K & 127;
    outp[idx] = (_Float16)W2[((size_t)n * DD + d) * DIM + Dcol];
}

// ---- Fused main kernel: gather -> GEMM1 -> GELU*mask -> GEMM2 -> residual ----
__global__ void __launch_bounds__(256) expandformer_main(
    const int*      __restrict__ x,
    const float*    __restrict__ embed,
    const _Float16* __restrict__ w1p,
    const _Float16* __restrict__ w2p,
    const int*      __restrict__ token_mask,
    float*          __restrict__ out)
{
    extern __shared__ char smem[];
    _Float16* actF16    = (_Float16*)(smem);                     // 32*2056*2 = 131584 B
    _Float16* hF16      = (_Float16*)(smem + 131584);            // 32*520*2  =  33280 B
    float*    maskScale = (float*)   (smem + 131584 + 33280);    // 32*16*4   =   2048 B
    int*      tokLDS    = (int*)     (smem + 131584 + 33280 + 2048); // 128 B

    const int tid      = threadIdx.x;
    const int wave     = tid >> 5;
    const int lane     = tid & 31;
    const int tileBase = blockIdx.x * MTILE;

    // ---- Phase A: tokens, masks, embedding gather (coalesced) ----
    if (tid < MTILE) tokLDS[tid] = x[tileBase + tid];
    __syncthreads();

    for (int i = tid; i < MTILE * ND; i += 256) {
        int row = i >> 4, n = i & 15;
        maskScale[i] = 0.1f * (float)token_mask[(size_t)tokLDS[row] * ND + n];
    }
    for (int q = tid; q < MTILE * (DIM / 4); q += 256) {
        int row = q >> 7;           // DIM/4 = 128 float4 per row
        int c4  = q & 127;
        const float4 v = *(const float4*)(embed + (size_t)tokLDS[row] * DIM + c4 * 4);
        v4h p = { (_Float16)v.x, (_Float16)v.y, (_Float16)v.z, (_Float16)v.w };
        *(v4h*)(hF16 + row * HSTR + c4 * 4) = p;
    }
    __syncthreads();

    const int rbase = (lane >= 16) ? 8 : 0;   // C-fragment row base
    const int ncol  = lane & 15;              // C/B column within 16-wide tile
    const int aoff0 = (lane >= 16) ? 8 : 0;   // A-fragment first K chunk base
    const int arow  = lane & 15;              // A-fragment row within M-tile

    // ---- Phase B: GEMM1 (h @ W1flat), GELU, mask, -> actF16 ----
    // wave owns hidden columns [256w, 256w+256); process 4 N-tiles at a time so
    // each pair of A-fragment loads feeds 8 WMMAs.
    for (int ng = 0; ng < 4; ++ng) {
        const int ntbase = wave * 16 + ng * 4;
        v8f acc[4][2] = {};
#pragma unroll 2
        for (int kt = 0; kt < 16; ++kt) {
            union { v16h v; v8h h[2]; } a[2];
#pragma unroll
            for (int mt = 0; mt < 2; ++mt) {
                const _Float16* ap = hF16 + (mt * 16 + arow) * HSTR + kt * 32 + aoff0;
                a[mt].h[0] = *(const v8h*)(ap);
                a[mt].h[1] = *(const v8h*)(ap + 16);
            }
#pragma unroll
            for (int t = 0; t < 4; ++t) {
                v16h b = *(const v16h*)(w1p + (size_t)((ntbase + t) * 16 + kt) * 512
                                        + lane * 16);
                acc[t][0] = __builtin_amdgcn_wmma_f32_16x16x32_f16(
                    false, a[0].v, false, b, (short)0, acc[t][0], false, false);
                acc[t][1] = __builtin_amdgcn_wmma_f32_16x16x32_f16(
                    false, a[1].v, false, b, (short)0, acc[t][1], false, false);
            }
        }
#pragma unroll
        for (int t = 0; t < 4; ++t) {
            const int ntile = ntbase + t;
            const int dom   = ntile >> 3;     // 8 ntiles per 128-wide domain
#pragma unroll
            for (int mt = 0; mt < 2; ++mt) {
#pragma unroll
                for (int r = 0; r < 8; ++r) {
                    const int row = mt * 16 + rbase + r;
                    float g = gelu_fast(acc[t][mt][r]);
                    actF16[row * ASTR + ntile * 16 + ncol] =
                        (_Float16)(g * maskScale[row * ND + dom]);
                }
            }
        }
    }
    __syncthreads();

    // ---- Phase C: GEMM2 (actm @ W2flat) + residual, store f32 ----
    // wave owns output columns [64w, 64w+64): all 4 N-tiles in one register-blocked pass.
    {
        const int ntbase2 = wave * 4;
        v8f acc[4][2] = {};
#pragma unroll 2
        for (int kt = 0; kt < 64; ++kt) {
            union { v16h v; v8h h[2]; } a[2];
#pragma unroll
            for (int mt = 0; mt < 2; ++mt) {
                const _Float16* ap = actF16 + (mt * 16 + arow) * ASTR + kt * 32 + aoff0;
                a[mt].h[0] = *(const v8h*)(ap);
                a[mt].h[1] = *(const v8h*)(ap + 16);
            }
#pragma unroll
            for (int t = 0; t < 4; ++t) {
                v16h b = *(const v16h*)(w2p + (size_t)((ntbase2 + t) * 64 + kt) * 512
                                        + lane * 16);
                acc[t][0] = __builtin_amdgcn_wmma_f32_16x16x32_f16(
                    false, a[0].v, false, b, (short)0, acc[t][0], false, false);
                acc[t][1] = __builtin_amdgcn_wmma_f32_16x16x32_f16(
                    false, a[1].v, false, b, (short)0, acc[t][1], false, false);
            }
        }
#pragma unroll
        for (int t = 0; t < 4; ++t) {
#pragma unroll
            for (int mt = 0; mt < 2; ++mt) {
#pragma unroll
                for (int r = 0; r < 8; ++r) {
                    const int row = mt * 16 + rbase + r;
                    const int col = (ntbase2 + t) * 16 + ncol;
                    const int tk  = tokLDS[row];
                    float hval = embed[(size_t)tk * DIM + col];   // exact f32 residual
                    out[(size_t)(tileBase + row) * DIM + col] = hval + acc[t][mt][r];
                }
            }
        }
    }
}

extern "C" void kernel_launch(void* const* d_in, const int* in_sizes, int n_in,
                              void* d_out, int out_size, void* d_ws, size_t ws_size,
                              hipStream_t stream) {
    const int*   xin   = (const int*)  d_in[0];   // [8,2048] int32
    const float* embed = (const float*)d_in[1];   // [32000,512] f32
    const float* W1    = (const float*)d_in[2];   // [16,512,128] f32
    const float* W2    = (const float*)d_in[3];   // [16,128,512] f32
    const int*   tmask = (const int*)  d_in[4];   // [32000,16] int32
    float*       outp  = (float*)d_out;           // [8,2048,512] f32

    _Float16* w1p = (_Float16*)d_ws;                      // 512*2048 halfs = 2 MB
    _Float16* w2p = w1p + (size_t)512 * HID;              // 2048*512 halfs = 2 MB

    pack_w1<<<(512 * HID) / 256, 256, 0, stream>>>(W1, w1p);
    pack_w2<<<(HID * DIM) / 256, 256, 0, stream>>>(W2, w2p);

    const size_t shmem = 131584 + 33280 + 2048 + 128;     // 167040 B < 320 KB/WGP
    expandformer_main<<<NTOK / MTILE, 256, shmem, stream>>>(
        xin, embed, w1p, w2p, tmask, outp);
}
